// MiMoV2Flash_2164663517574
// MI455X (gfx1250) — matrix operand order repaired and checked by reading, not compiled, not run on hardware
//
#include <hip/hip_runtime.h>
#include <hip/hip_bf16.h>
#include <cstdint>

// ---------------- model constants ----------------
#define DIM 512
#define NH 8
#define NKV 2
#define HD 64
#define E_EXP 16
#define EH 512
#define VVOCAB 32000
#define BATCH 2
#define SEQ 1024
#define TOK (BATCH * SEQ)
#define ROPE_DIM 32

// ---------------- WMMA / TDM types ----------------
typedef __attribute__((ext_vector_type(16))) __bf16 v16bf;
typedef __attribute__((ext_vector_type(8)))  float  v8f;
typedef __attribute__((ext_vector_type(4)))  unsigned int u32x4;
typedef __attribute__((ext_vector_type(8)))  int i32x8;
typedef __attribute__((ext_vector_type(4)))  int i32x4;

union Frag16 {
    v16bf v;
    uint4 q[2];
};

__device__ __forceinline__ unsigned short f2bf(float f) {
    unsigned int u = __float_as_uint(f);
    unsigned int r = (u + 0x7FFFu + ((u >> 16) & 1u)) >> 16;
    return (unsigned short)r;
}

// Issue one TDM load: 32(M-rows) x 32(K-halves) bf16 tile of a row-major
// [M x K] matrix into LDS at lds_off.  D# per CDNA5 ISA ch.8.
__device__ __forceinline__ void tdm_load_tile32x32(const unsigned short* gptr,
                                                   unsigned lds_off,
                                                   int K, int M) {
    unsigned long long ga = (unsigned long long)(uintptr_t)gptr;
    u32x4 g0;
    g0[0] = 1u;                                   // count=1 valid descriptor
    g0[1] = lds_off;                              // lds_addr (bytes)
    g0[2] = (unsigned)(ga & 0xFFFFFFFFull);       // global_addr[31:0]
    g0[3] = (unsigned)((ga >> 32) & 0x1FFFFFFull) // global_addr[56:32]
          | 0x80000000u;                          // type=2 ("image")
    i32x8 g1;
    g1[0] = 1 << 16;                              // data_size=1 (2 bytes), wg_mask=0
    g1[1] = (K & 0xFFFF) << 16;                   // tensor_dim0[15:0]
    g1[2] = ((unsigned)K >> 16) | ((M & 0xFFFF) << 16);   // dim0[31:16] | dim1[15:0]
    g1[3] = (((unsigned)M >> 16) & 0xFFFF) | (32 << 16);  // dim1[31:16] | tile_dim0=32
    g1[4] = 32;                                   // tile_dim1=32, tile_dim2=0
    g1[5] = K;                                    // tensor_dim0_stride[31:0]
    g1[6] = 0;                                    // stride0[47:32] | stride1[15:0]
    g1[7] = 0;
    i32x4 z4 = {0, 0, 0, 0};
    i32x8 z8 = {0, 0, 0, 0, 0, 0, 0, 0};
    __builtin_amdgcn_tensor_load_to_lds(g0, g1, z4, z4, z8, 0);
}

// ---------------- embedding gather + (reference-style) rope ----------------
__global__ void embed_rope_kernel(const int* __restrict__ ids,
                                  const float* __restrict__ emb,
                                  float* __restrict__ X) {
    int tok = blockIdx.x;                 // 0..TOK-1
    int pos = tok % SEQ;
    int id = ids[tok];
    float* xr = X + (size_t)tok * DIM;
    const float* er = emb + (size_t)id * DIM;
    for (int i = threadIdx.x; i < DIM; i += blockDim.x) xr[i] = er[i];
    __syncthreads();
    if (threadIdx.x < ROPE_DIM / 2) {
        int j = threadIdx.x;
        // theta[t] = 10000^(-(2t)/32); reference multiplies raw angles (no cos/sin).
        float tc = __powf(10000.0f, -(float)(((2 * j) % 16) * 2) / 32.0f);
        float ts = __powf(10000.0f, -(float)(((2 * j + 1) % 16) * 2) / 32.0f);
        float c = (float)pos * tc;
        float sn = (float)pos * ts;
        float xe = xr[2 * j], xo = xr[2 * j + 1];
        xr[2 * j]     = xe * c - xo * sn;
        xr[2 * j + 1] = xo * c + xe * sn;
    }
}

// ---------------- rmsnorm: one block per row ----------------
__global__ void rmsnorm_kernel(const float* __restrict__ X,
                               const float* __restrict__ w,
                               float* __restrict__ outF,
                               unsigned short* __restrict__ out16) {
    __shared__ float red[256];
    int n = blockIdx.x;
    const float* xr = X + (size_t)n * DIM;
    float s = 0.0f;
    for (int i = threadIdx.x; i < DIM; i += 256) { float v = xr[i]; s += v * v; }
    red[threadIdx.x] = s;
    __syncthreads();
    for (int st = 128; st > 0; st >>= 1) {
        if (threadIdx.x < st) red[threadIdx.x] += red[threadIdx.x + st];
        __syncthreads();
    }
    float inv = rsqrtf(red[0] / (float)DIM + 1e-6f);
    for (int i = threadIdx.x; i < DIM; i += 256) {
        float v = xr[i] * inv * w[i];
        if (outF)  outF[(size_t)n * DIM + i] = v;
        if (out16) out16[(size_t)n * DIM + i] = f2bf(v);
    }
}

// ---------------- f32 (K,N) -> bf16 (N,K) transpose-convert ----------------
__global__ void transpose_cvt_kernel(const float* __restrict__ src,
                                     unsigned short* __restrict__ dst,
                                     int K, int N) {
    size_t idx = (size_t)blockIdx.x * 256 + threadIdx.x;
    if (idx >= (size_t)K * N) return;
    int n = (int)(idx % N);
    int k = (int)(idx / N);
    dst[(size_t)n * K + k] = f2bf(src[idx]);
}

__global__ void cvt_bf16_kernel(const float* __restrict__ src,
                                unsigned short* __restrict__ dst, size_t n) {
    size_t idx = (size_t)blockIdx.x * 256 + threadIdx.x;
    if (idx < n) dst[idx] = f2bf(src[idx]);
}

// ---------------- WMMA GEMM: C[M,N] = A[M,K] * Bt[N,K]^T (+bias,+silu,+resid) ----
// Block tile: 32M x 128N, 4 waves; each wave: 32M x 32N = 4 accumulators.
// A tile (32x32 bf16) streamed into double-buffered LDS by the Tensor Data
// Mover (wave 0 issues, TENSORcnt-pipelined); B fragments loaded from the
// pre-transposed bf16 weights and reused across both M sub-tiles.
__global__ void __launch_bounds__(128)
wmma_gemm_bf16_kernel(const unsigned short* __restrict__ A,   // M x K bf16
                      const unsigned short* __restrict__ Bt,  // N x K bf16
                      const float* __restrict__ bias,         // N or null
                      const float* __restrict__ resid,        // M x N or null
                      float* __restrict__ Cf,                 // M x N or null
                      unsigned short* __restrict__ C16,       // M x N or null
                      int M, int N, int K, int act) {
    __shared__ __align__(16) unsigned short As[2][32 * 32];
    const int lane = threadIdx.x & 31;
    const int wave = threadIdx.x >> 5;
    const int m0 = blockIdx.y * 32;
    const int n0 = blockIdx.x * 128 + wave * 32;
    const int r16 = lane & 15;
    const int hi = lane >> 4;

    v8f acc00 = {}, acc01 = {}, acc10 = {}, acc11 = {};
    const unsigned short* brow0 = Bt + (size_t)(n0 + r16) * K + hi * 8;
    const unsigned short* brow1 = Bt + (size_t)(n0 + 16 + r16) * K + hi * 8;
    const unsigned short* arow = A + (size_t)m0 * K;
    unsigned lds0 = (unsigned)(uintptr_t)&As[0][0];
    unsigned lds1 = (unsigned)(uintptr_t)&As[1][0];

    if (wave == 0) tdm_load_tile32x32(arow, lds0, K, M);

    int cur = 0;
    for (int kk = 0; kk < K; kk += 32) {
        if (wave == 0) {
            if (kk + 32 < K) {
                tdm_load_tile32x32(arow + kk + 32, cur ? lds0 : lds1, K, M);
                __builtin_amdgcn_s_wait_tensorcnt(1);   // buf[cur] complete
            } else {
                __builtin_amdgcn_s_wait_tensorcnt(0);
            }
        }
        __syncthreads();                                 // buf[cur] visible to all

        const unsigned short* as = &As[cur][0];
        Frag16 a0, a1, b0, b1;
        {   // ISA 16-bit A layout: lane<16 -> K 0..7 & 16..23 ; lane>=16 -> 8..15 & 24..31
            const unsigned short* p0 = &as[r16 * 32 + hi * 8];
            a0.q[0] = *(const uint4*)(p0);
            a0.q[1] = *(const uint4*)(p0 + 16);
            const unsigned short* p1 = &as[(16 + r16) * 32 + hi * 8];
            a1.q[0] = *(const uint4*)(p1);
            a1.q[1] = *(const uint4*)(p1 + 16);
        }
        {   // ISA 16-bit B layout: lane<16 -> K 0..7 & 16..23 ; lane>=16 -> 8..15 & 24..31
            const unsigned short* bp0 = brow0 + kk;
            b0.q[0] = *(const uint4*)(bp0);
            b0.q[1] = *(const uint4*)(bp0 + 16);
            const unsigned short* bp1 = brow1 + kk;
            b1.q[0] = *(const uint4*)(bp1);
            b1.q[1] = *(const uint4*)(bp1 + 16);
        }
        if (kk + 32 < K) {
            __builtin_prefetch(brow0 + kk + 32, 0, 1);
            __builtin_prefetch(brow1 + kk + 32, 0, 1);
        }
        acc00 = __builtin_amdgcn_wmma_f32_16x16x32_bf16(false, a0.v, false, b0.v, (short)0, acc00, false, false);
        acc01 = __builtin_amdgcn_wmma_f32_16x16x32_bf16(false, a0.v, false, b1.v, (short)0, acc01, false, false);
        acc10 = __builtin_amdgcn_wmma_f32_16x16x32_bf16(false, a1.v, false, b0.v, (short)0, acc10, false, false);
        acc11 = __builtin_amdgcn_wmma_f32_16x16x32_bf16(false, a1.v, false, b1.v, (short)0, acc11, false, false);
        __syncthreads();                                 // done reading buf[cur]
        cur ^= 1;
    }

    // epilogue: 4 tiles
    v8f* accs[4] = { &acc00, &acc01, &acc10, &acc11 };
#pragma unroll
    for (int t = 0; t < 4; ++t) {
        int mt = t >> 1, nt = t & 1;
        int n = n0 + nt * 16 + r16;
        int mb = m0 + mt * 16 + hi * 8;
        float bv = bias ? bias[n] : 0.0f;
        v8f& acc = *accs[t];
#pragma unroll
        for (int r = 0; r < 8; ++r) {
            int m = mb + r;
            float v = acc[r] + bv;
            if (act) v = v / (1.0f + __expf(-v));   // silu
            if (resid) v += resid[(size_t)m * N + n];
            if (Cf)  Cf[(size_t)m * N + n] = v;
            if (C16) C16[(size_t)m * N + n] = f2bf(v);
        }
    }
}

// ---------------- attention: one wave per (b,h,q), online softmax with sink ----
__global__ void __launch_bounds__(128)
attention_kernel(const float* __restrict__ Q,   // TOK x (NH*HD)
                 const float* __restrict__ Kc,  // TOK x (NKV*HD)
                 const float* __restrict__ Vc,  // TOK x (NKV*HD)
                 const float* __restrict__ sinkp,
                 float* __restrict__ CTX,       // TOK x (NH*HD)
                 int window) {
    const int lane = threadIdx.x & 31;
    const int wave = threadIdx.x >> 5;
    int g = blockIdx.x * 4 + wave;       // 0..B*NH*S-1
    int i = g % SEQ;
    int h = (g / SEQ) % NH;
    int b = g / (SEQ * NH);
    int row = b * SEQ + i;
    int kv = h >> 2;                      // NH/NKV = 4

    const float* qr = Q + (size_t)row * (NH * HD) + h * HD;
    float q0 = qr[lane], q1 = qr[lane + 32];
    const float scale = 0.125f;           // 1/sqrt(64)
    float sink = sinkp[0];

    int jstart = i - (window - 1);
    if (jstart < 0) jstart = 0;

    float m = sink;
    float l = 1.0f;                       // sink term exp(sink - m) at m==sink
    float a0 = 0.0f, a1 = 0.0f;

    for (int j = jstart; j <= i; ++j) {
        int krow = b * SEQ + j;
        const float* kr = Kc + (size_t)krow * (NKV * HD) + kv * HD;
        float part = q0 * kr[lane] + q1 * kr[lane + 32];
        for (int off = 16; off > 0; off >>= 1) part += __shfl_xor(part, off, 32);
        float s = part * scale;
        float mn = fmaxf(m, s);
        float r = __expf(m - mn);
        float p = __expf(s - mn);
        const float* vr = Vc + (size_t)krow * (NKV * HD) + kv * HD;
        l = l * r + p;
        a0 = a0 * r + p * vr[lane];
        a1 = a1 * r + p * vr[lane + 32];
        m = mn;
    }
    float* o = CTX + (size_t)row * (NH * HD) + h * HD;
    o[lane]      = a0 / l;
    o[lane + 32] = a1 / l;
}

// ---------------- router: top-2 + softmax + aux colsum, one wave per token ----
__global__ void router_kernel(const float* __restrict__ XNF,
                              const float* __restrict__ rw,   // DIM x E
                              const float* __restrict__ rb,   // E
                              float* __restrict__ comb,       // TOK x E
                              float* __restrict__ colsum) {   // E
    int n = blockIdx.x;
    int lane = threadIdx.x;
    float logit = -1e30f;
    if (lane < E_EXP) {
        const float* xr = XNF + (size_t)n * DIM;
        float acc = 0.0f;
        for (int k = 0; k < DIM; ++k) acc += xr[k] * rw[k * E_EXP + lane];
        logit = (acc + rb[lane]) * 10.0f;   // / 0.1
    }
    // top1 (jax tie-break: smaller index wins)
    float v = logit; int idx = lane;
    for (int off = 8; off > 0; off >>= 1) {
        float ov = __shfl_xor(v, off, 16);
        int   oi = __shfl_xor(idx, off, 16);
        if (ov > v || (ov == v && oi < idx)) { v = ov; idx = oi; }
    }
    float t1v = v; int t1i = idx;
    float v2 = (lane == t1i) ? -1e30f : logit; int idx2 = lane;
    for (int off = 8; off > 0; off >>= 1) {
        float ov = __shfl_xor(v2, off, 16);
        int   oi = __shfl_xor(idx2, off, 16);
        if (ov > v2 || (ov == v2 && oi < idx2)) { v2 = ov; idx2 = oi; }
    }
    float w1 = 1.0f / (1.0f + __expf(v2 - t1v));
    float w2 = 1.0f - w1;
    // full softmax for aux
    float p = (lane < E_EXP) ? __expf(logit - t1v) : 0.0f;
    float ps = p;
    for (int off = 8; off > 0; off >>= 1) ps += __shfl_xor(ps, off, 16);
    if (lane < E_EXP) {
        atomicAdd(&colsum[lane], p / ps);
        float c = (lane == t1i) ? w1 : ((lane == idx2) ? w2 : 0.0f);
        comb[(size_t)n * E_EXP + lane] = c;
    }
}

__global__ void aux_kernel(const float* __restrict__ colsum, float* __restrict__ auxacc) {
    if (threadIdx.x == 0 && blockIdx.x == 0) {
        float s = 0.0f;
        for (int e = 0; e < E_EXP; ++e) s += colsum[e] * colsum[e];
        auxacc[0] += s * (1e-5f / (float)E_EXP);
    }
}

__global__ void moe_accum_kernel(float* __restrict__ X, const float* __restrict__ Y,
                                 const float* __restrict__ comb, int e) {
    size_t idx = (size_t)blockIdx.x * 256 + threadIdx.x;
    if (idx >= (size_t)TOK * DIM) return;
    size_t n = idx / DIM;
    X[idx] += comb[n * E_EXP + e] * Y[idx];
}

__global__ void write_aux_kernel(float* __restrict__ dst, const float* __restrict__ auxacc) {
    if (threadIdx.x == 0 && blockIdx.x == 0) dst[0] = auxacc[0];
}

// =====================================================================
extern "C" void kernel_launch(void* const* d_in, const int* in_sizes, int n_in,
                              void* d_out, int out_size, void* d_ws, size_t ws_size,
                              hipStream_t stream) {
    (void)in_sizes; (void)n_in; (void)out_size; (void)ws_size;

    // ---- walk inputs in setup_inputs() dict order ----
    int ci = 0;
    const int*   ids = (const int*)d_in[ci++];
    const float* emb = (const float*)d_in[ci++];
    struct Blk {
        const float *norm1, *norm2, *q_w, *q_b, *k_w, *k_b, *v_w, *v_b, *o_w, *o_b, *sink;
        const float *w1, *b1, *w2, *b2;                         // dense
        const float *router_w, *router_b, *W1, *b1e, *W2, *b2e; // moe
    };
    const int kinds[7]   = {0, 1, 1, 1, 1, 1, 1};   // 0=dense 1=moe
    const int windows[7] = {10000, 64, 64, 10000, 64, 64, 10000};
    Blk blk[7];
    for (int L = 0; L < 7; ++L) {
        Blk& b = blk[L];
        b.norm1 = (const float*)d_in[ci++]; b.norm2 = (const float*)d_in[ci++];
        b.q_w = (const float*)d_in[ci++];   b.q_b = (const float*)d_in[ci++];
        b.k_w = (const float*)d_in[ci++];   b.k_b = (const float*)d_in[ci++];
        b.v_w = (const float*)d_in[ci++];   b.v_b = (const float*)d_in[ci++];
        b.o_w = (const float*)d_in[ci++];   b.o_b = (const float*)d_in[ci++];
        b.sink = (const float*)d_in[ci++];
        if (kinds[L] == 0) {
            b.w1 = (const float*)d_in[ci++]; b.b1 = (const float*)d_in[ci++];
            b.w2 = (const float*)d_in[ci++]; b.b2 = (const float*)d_in[ci++];
        } else {
            b.router_w = (const float*)d_in[ci++]; b.router_b = (const float*)d_in[ci++];
            b.W1 = (const float*)d_in[ci++]; b.b1e = (const float*)d_in[ci++];
            b.W2 = (const float*)d_in[ci++]; b.b2e = (const float*)d_in[ci++];
        }
    }
    const float* norm_f = (const float*)d_in[ci++];

    // ---- workspace carve-up ----
    char* ws = (char*)d_ws;
    size_t off = 0;
    auto alloc = [&](size_t bytes) -> void* {
        void* p = ws + off;
        off += (bytes + 255) & ~(size_t)255;
        return p;
    };
    float* X    = (float*)alloc((size_t)TOK * DIM * 4);
    float* XNF  = (float*)alloc((size_t)TOK * DIM * 4);
    unsigned short* XN16 = (unsigned short*)alloc((size_t)TOK * DIM * 2);
    float* Qb   = (float*)alloc((size_t)TOK * (NH * HD) * 4);
    float* Kb   = (float*)alloc((size_t)TOK * (NKV * HD) * 4);
    float* Vb   = (float*)alloc((size_t)TOK * (NKV * HD) * 4);
    float* CTX  = (float*)alloc((size_t)TOK * DIM * 4);
    unsigned short* CTX16 = (unsigned short*)alloc((size_t)TOK * DIM * 2);
    unsigned short* H16   = (unsigned short*)alloc((size_t)TOK * 2048 * 2);
    float* Yb   = (float*)alloc((size_t)TOK * DIM * 4);
    unsigned short* WT    = (unsigned short*)alloc((size_t)2048 * 512 * 2);
    unsigned short* EMB16 = (unsigned short*)alloc((size_t)VVOCAB * DIM * 2);
    float* COMB   = (float*)alloc((size_t)TOK * E_EXP * 4);
    float* COLSUM = (float*)alloc(E_EXP * 4);
    float* AUXACC = (float*)alloc(4);

    auto gemm = [&](const unsigned short* A, const unsigned short* Bt, const float* bias,
                    const float* resid, float* Cf, unsigned short* C16,
                    int M, int N, int K, int act) {
        dim3 grid(N / 128, M / 32);
        wmma_gemm_bf16_kernel<<<grid, 128, 0, stream>>>(A, Bt, bias, resid, Cf, C16, M, N, K, act);
    };
    auto trw = [&](const float* w, int K, int N) {
        size_t tot = (size_t)K * N;
        transpose_cvt_kernel<<<(int)((tot + 255) / 256), 256, 0, stream>>>(w, WT, K, N);
    };

    (void)hipMemsetAsync(AUXACC, 0, 4, stream);

    // ---- embedding + rope ----
    embed_rope_kernel<<<TOK, 256, 0, stream>>>(ids, emb, X);

    // ---- layers ----
    for (int L = 0; L < 7; ++L) {
        const Blk& b = blk[L];
        // attention block
        rmsnorm_kernel<<<TOK, 256, 0, stream>>>(X, b.norm1, nullptr, XN16);
        trw(b.q_w, DIM, NH * HD);
        gemm(XN16, WT, b.q_b, nullptr, Qb, nullptr, TOK, NH * HD, DIM, 0);
        trw(b.k_w, DIM, NKV * HD);
        gemm(XN16, WT, b.k_b, nullptr, Kb, nullptr, TOK, NKV * HD, DIM, 0);
        trw(b.v_w, DIM, NKV * HD);
        gemm(XN16, WT, b.v_b, nullptr, Vb, nullptr, TOK, NKV * HD, DIM, 0);
        attention_kernel<<<(BATCH * NH * SEQ) / 4, 128, 0, stream>>>(Qb, Kb, Vb, b.sink, CTX, windows[L]);
        cvt_bf16_kernel<<<(TOK * DIM) / 256, 256, 0, stream>>>(CTX, CTX16, (size_t)TOK * DIM);
        trw(b.o_w, NH * HD, DIM);
        gemm(CTX16, WT, b.o_b, X, X, nullptr, TOK, DIM, NH * HD, 0);   // x += attn

        // ffn block
        rmsnorm_kernel<<<TOK, 256, 0, stream>>>(X, b.norm2, XNF, XN16);
        if (kinds[L] == 0) {
            trw(b.w1, DIM, 2048);
            gemm(XN16, WT, b.b1, nullptr, nullptr, H16, TOK, 2048, DIM, 1);   // silu
            trw(b.w2, 2048, DIM);
            gemm(H16, WT, b.b2, X, X, nullptr, TOK, DIM, 2048, 0);            // x += ffn
        } else {
            (void)hipMemsetAsync(COLSUM, 0, E_EXP * 4, stream);
            router_kernel<<<TOK, 32, 0, stream>>>(XNF, b.router_w, b.router_b, COMB, COLSUM);
            aux_kernel<<<1, 1, 0, stream>>>(COLSUM, AUXACC);
            for (int e = 0; e < E_EXP; ++e) {
                trw(b.W1 + (size_t)e * DIM * EH, DIM, EH);
                gemm(XN16, WT, b.b1e + (size_t)e * EH, nullptr, nullptr, H16, TOK, EH, DIM, 1);
                trw(b.W2 + (size_t)e * EH * DIM, EH, DIM);
                gemm(H16, WT, b.b2e + (size_t)e * DIM, nullptr, Yb, nullptr, TOK, DIM, EH, 0);
                moe_accum_kernel<<<(TOK * DIM) / 256, 256, 0, stream>>>(X, Yb, COMB, e);
            }
        }
    }

    // ---- final norm + logits ----
    rmsnorm_kernel<<<TOK, 256, 0, stream>>>(X, norm_f, nullptr, XN16);
    cvt_bf16_kernel<<<((size_t)VVOCAB * DIM + 255) / 256, 256, 0, stream>>>(emb, EMB16, (size_t)VVOCAB * DIM);
    float* logits = (float*)d_out;
    gemm(XN16, EMB16, nullptr, nullptr, logits, nullptr, TOK, VVOCAB, DIM, 0);
    write_aux_kernel<<<1, 1, 0, stream>>>(logits + (size_t)TOK * VVOCAB, AUXACC);
}
